// Attention_87668872446719
// MI455X (gfx1250) — matrix operand order repaired and checked by reading, not compiled, not run on hardware
//
#include <hip/hip_runtime.h>
#include <hip/hip_bf16.h>
#include <stdint.h>

// ---------------------------------------------------------------------------
// Shapes
// ---------------------------------------------------------------------------
#define BB    4
#define CIN   3
#define HH    224
#define WW    224
#define PP    16
#define DIM   256
#define HEADS 8
#define HD    32          // DIM / HEADS
#define HP    14          // H / P
#define NN    196         // HP*HP patches
#define NP    224         // N padded to multiple of 32 (WMMA K step)
#define PIX   256         // P*P pixels per patch
#define COLS1 8192        // HD * PIX  (inner dim per head of GEMM-1)

typedef __attribute__((ext_vector_type(16))) __bf16 v16bf;
typedef __attribute__((ext_vector_type(8)))  float  v8f;

union FragB16 { v16bf v; uint4 u[2]; };

// ---------------------------------------------------------------------------
// Workspace layout (bytes)
// ---------------------------------------------------------------------------
static constexpr size_t OFF_XE   = 0;                          // [B,N,DIM] f32      802816
static constexpr size_t OFF_Q    = OFF_XE   + 802816;          // [B,H,N,HD] f32     802816
static constexpr size_t OFF_K    = OFF_Q    + 802816;          // [B,H,N,HD] f32     802816
static constexpr size_t OFF_ATTN = OFF_K    + 802816;          // [B*H,NP,NP] bf16   3211264
static constexpr size_t OFF_VBT  = OFF_ATTN + 3211264;         // [B,DIM,PIX,NP] bf16 117440512
static constexpr size_t OFF_OUTP = OFF_VBT  + 117440512;       // [B,N,PIX,DIM] bf16 102760448
static constexpr size_t OFF_PW   = OFF_OUTP + 102760448;       // [DIM,DIM] bf16     131072

// ---------------------------------------------------------------------------
// K0a: patch embed  xe[b,n,c] = patch_b[c] + <x patch, patch_w[c]>
// ---------------------------------------------------------------------------
__global__ void k_patch_embed(const float* __restrict__ x, const float* __restrict__ pw,
                              const float* __restrict__ pb, float* __restrict__ xe) {
  int idx = blockIdx.x * blockDim.x + threadIdx.x;           // B*N*DIM
  if (idx >= BB * NN * DIM) return;
  int c = idx % DIM;
  int n = (idx / DIM) % NN;
  int b = idx / (DIM * NN);
  int hp = n / HP, wp = n % HP;
  float acc = pb[c];
  for (int ci = 0; ci < CIN; ++ci)
    for (int ki = 0; ki < PP; ++ki) {
      const float* xr = x  + (((size_t)b * CIN + ci) * HH + hp * PP + ki) * WW + wp * PP;
      const float* wr = pw + ((c * CIN + ci) * PP + ki) * PP;
#pragma unroll
      for (int kj = 0; kj < PP; ++kj) acc += xr[kj] * wr[kj];
    }
  xe[((size_t)b * NN + n) * DIM + c] = acc;
}

// ---------------------------------------------------------------------------
// K0b: qk projection  -> q,k [B,H,N,HD] f32
// ---------------------------------------------------------------------------
__global__ void k_qk(const float* __restrict__ xe, const float* __restrict__ qkw,
                     float* __restrict__ q, float* __restrict__ k) {
  int idx = blockIdx.x * blockDim.x + threadIdx.x;           // B*N*2*DIM
  if (idx >= BB * NN * 2 * DIM) return;
  int oj = idx % (2 * DIM);
  int n  = (idx / (2 * DIM)) % NN;
  int b  = idx / (2 * DIM * NN);
  const float* xr = xe  + ((size_t)b * NN + n) * DIM;
  const float* wr = qkw + (size_t)oj * DIM;
  float acc = 0.f;
#pragma unroll 8
  for (int d = 0; d < DIM; ++d) acc += xr[d] * wr[d];
  int two = oj >> 8, h = (oj >> 5) & 7, dd = oj & 31;
  float* dst = two ? k : q;
  dst[(((size_t)b * HEADS + h) * NN + n) * HD + dd] = acc;
}

// ---------------------------------------------------------------------------
// K2: scores + softmax, one block per (bh, padded row), write bf16, zero pad
// ---------------------------------------------------------------------------
__global__ void k_attn_softmax(const float* __restrict__ q, const float* __restrict__ k,
                               __bf16* __restrict__ attn) {
  int row = blockIdx.x;                  // 0 .. B*H*NP-1
  int n   = row % NP;
  int bh  = row / NP;
  int tid = threadIdx.x;                 // 256 threads
  __bf16* arow = attn + (size_t)row * NP;
  if (n >= NN) { if (tid < NP) arow[tid] = (__bf16)0.0f; return; }

  __shared__ float qs[HD];
  __shared__ float red[256];
  if (tid < HD) qs[tid] = q[((size_t)bh * NN + n) * HD + tid];
  __syncthreads();

  float val = 0.f, s = -1e30f;
  if (tid < NN) {
    const float* kr = k + ((size_t)bh * NN + tid) * HD;
    float acc = 0.f;
#pragma unroll
    for (int d = 0; d < HD; ++d) acc += qs[d] * kr[d];
    val = acc * 0.17677669529663687f;    // 32^-0.5
    s = val;
  }
  red[tid] = s; __syncthreads();
  for (int off = 128; off > 0; off >>= 1) {
    if (tid < off) red[tid] = fmaxf(red[tid], red[tid + off]);
    __syncthreads();
  }
  float mx = red[0]; __syncthreads();
  float e = (tid < NN) ? __expf(val - mx) : 0.f;
  red[tid] = e; __syncthreads();
  for (int off = 128; off > 0; off >>= 1) {
    if (tid < off) red[tid] += red[tid + off];
    __syncthreads();
  }
  float inv = 1.0f / red[0];
  if (tid < NP) arow[tid] = (tid < NN) ? (__bf16)(e * inv) : (__bf16)0.0f;
}

// ---------------------------------------------------------------------------
// K1: conv3x3 V, packed directly as bf16 B-operand: vbt[b, c, p, m_pad]
// (m contiguous so WMMA B-fragments are two 16B loads per lane)
// ---------------------------------------------------------------------------
__global__ void k_conv_v(const float* __restrict__ x, const float* __restrict__ vw,
                         const float* __restrict__ vb, __bf16* __restrict__ vbt) {
  size_t idx = (size_t)blockIdx.x * blockDim.x + threadIdx.x;
  if (idx >= (size_t)BB * DIM * PIX * NP) return;
  int mp = idx % NP;
  size_t r = idx / NP;
  int p = r % PIX; r /= PIX;
  int c = r % DIM;
  int b = r / DIM;
  if (mp >= NN) { vbt[idx] = (__bf16)0.0f; return; }
  int hp = mp / HP, wp = mp % HP;
  int y  = hp * PP + (p >> 4);
  int xx = wp * PP + (p & 15);
  float acc = vb[c];
#pragma unroll
  for (int ci = 0; ci < CIN; ++ci)
#pragma unroll
    for (int dy = 0; dy < 3; ++dy) {
      int yy = y + dy - 1;
      if (yy < 0 || yy >= HH) continue;
#pragma unroll
      for (int dx = 0; dx < 3; ++dx) {
        int xc = xx + dx - 1;
        if (xc < 0 || xc >= WW) continue;
        acc += x[(((size_t)b * CIN + ci) * HH + yy) * WW + xc] *
               vw[((c * CIN + ci) * 3 + dy) * 3 + dx];
      }
    }
  vbt[idx] = (__bf16)acc;
}

// ---------------------------------------------------------------------------
// pack proj_w (1x1 conv weight) to bf16 [oc, c]
// ---------------------------------------------------------------------------
__global__ void k_pack_w(const float* __restrict__ w, __bf16* __restrict__ wb) {
  int i = blockIdx.x * blockDim.x + threadIdx.x;
  if (i < DIM * DIM) wb[i] = (__bf16)w[i];
}

// ---------------------------------------------------------------------------
// K3: out_pre = attn @ vu  (per (b,h): M=224pad, K=224pad, Ncols=8192)
// one wave computes a 32x64 tile: 2 A-frags x 4 B-frags -> 8 WMMA per K-step.
// Column enumeration j = p*32 + ch (NOT ch*256 + p): a 16-wide tile then has
// fixed pixel p and lane-varying channel -> D-tile stores hit consecutive c,
// the fastest axis of outp[b,n,p,c] (coalesced 64B spans instead of 32
// scattered sectors). vbt/outp layouts and K4 are unaffected.
// ---------------------------------------------------------------------------
__global__ void __launch_bounds__(256)
k_gemm_attn_v(const __bf16* __restrict__ attn,   // [B*H, NP, NP]
              const __bf16* __restrict__ vbt,    // [B, DIM, PIX, NP]
              __bf16* __restrict__ outp) {       // [B, NN, PIX, DIM]
  int bh   = blockIdx.x;                 // 0..31
  int b = bh >> 3, h = bh & 7;
  int wave = threadIdx.x >> 5;
  int lane = threadIdx.x & 31;
  int task = blockIdx.y * 8 + wave;      // 0..895
  int mgrp = task / 128;                 // 0..6  (7 groups of 2 M-tiles = 224 rows)
  int j0   = (task % 128) * 64;          // column base into 8192
  int lrow  = lane & 15;
  int lhalf = lane >> 4;
  const __bf16* A = attn + (size_t)bh * NP * NP;

  v8f acc[2][4] = {};
  int arow0 = mgrp * 32 + lrow;          // <= 223 < NP
  int arow1 = arow0 + 16;
  int koffA = lhalf * 8;
  int kbB   = lhalf * 8;

  for (int kk = 0; kk < NP / 32; ++kk) {
    int k0 = kk * 32;
    FragB16 fa0, fa1;
    fa0.u[0] = *(const uint4*)(A + (size_t)arow0 * NP + k0 + koffA);
    fa0.u[1] = *(const uint4*)(A + (size_t)arow0 * NP + k0 + 16 + koffA);
    fa1.u[0] = *(const uint4*)(A + (size_t)arow1 * NP + k0 + koffA);
    fa1.u[1] = *(const uint4*)(A + (size_t)arow1 * NP + k0 + 16 + koffA);
    if (kk + 1 < NP / 32) {
      __builtin_prefetch(A + (size_t)arow0 * NP + k0 + 32, 0, 1);
      __builtin_prefetch(A + (size_t)arow1 * NP + k0 + 32, 0, 1);
    }
#pragma unroll
    for (int t = 0; t < 4; ++t) {
      int j = j0 + t * 16 + lrow;        // j = p*HD + ch
      int c = h * HD + (j & (HD - 1));
      int p = j >> 5;
      const __bf16* bp = vbt + (((size_t)b * DIM + c) * PIX + p) * NP + k0 + kbB;
      FragB16 fb;
      fb.u[0] = *(const uint4*)(bp);
      fb.u[1] = *(const uint4*)(bp + 16);
      acc[0][t] = __builtin_amdgcn_wmma_f32_16x16x32_bf16(
          false, fa0.v, false, fb.v, (short)0, acc[0][t], false, false);
      acc[1][t] = __builtin_amdgcn_wmma_f32_16x16x32_bf16(
          false, fa1.v, false, fb.v, (short)0, acc[1][t], false, false);
    }
  }
#pragma unroll
  for (int mi = 0; mi < 2; ++mi) {
#pragma unroll
    for (int t = 0; t < 4; ++t) {
      int j = j0 + t * 16 + lrow;        // j = p*HD + ch
      int c = h * HD + (j & (HD - 1));
      int p = j >> 5;
#pragma unroll
      for (int v = 0; v < 8; ++v) {
        int n = mgrp * 32 + mi * 16 + v + lhalf * 8;
        if (n < NN)
          outp[(((size_t)b * NN + n) * PIX + p) * DIM + c] = (__bf16)acc[mi][t][v];
      }
    }
  }
}

// ---------------------------------------------------------------------------
// K4: final = proj_w @ out_pre + proj_b, folded to [B,DIM,H,W] fp32
// per b: M=256 (oc), K=256 (c), Ncols=50176 (n*256+p)
// one wave computes a 32x64 tile (2 M-tiles x 4 N-tiles, 8 WMMA per K-step)
// ---------------------------------------------------------------------------
__global__ void __launch_bounds__(256)
k_gemm_proj(const __bf16* __restrict__ pw,     // [DIM, DIM]
            const __bf16* __restrict__ outp,   // [B, NN, PIX, DIM]
            const float* __restrict__ pb,
            float* __restrict__ out) {         // [B, DIM, HH, WW]
  int b    = blockIdx.x;
  int wave = threadIdx.x >> 5;
  int lane = threadIdx.x & 31;
  int task = blockIdx.y * 8 + wave;      // 0..6271
  int mgrp = task / 784;                 // 0..7  (8 groups of 2 M-tiles = 256 oc)
  int j0   = (task % 784) * 64;          // 0..50112
  int lrow  = lane & 15;
  int lhalf = lane >> 4;

  v8f acc[2][4] = {};
  int arow0 = mgrp * 32 + lrow;
  int arow1 = arow0 + 16;
  int koffA = lhalf * 8;
  int kbB   = lhalf * 8;

  for (int kk = 0; kk < DIM / 32; ++kk) {
    int k0 = kk * 32;
    FragB16 fa0, fa1;
    fa0.u[0] = *(const uint4*)(pw + (size_t)arow0 * DIM + k0 + koffA);
    fa0.u[1] = *(const uint4*)(pw + (size_t)arow0 * DIM + k0 + 16 + koffA);
    fa1.u[0] = *(const uint4*)(pw + (size_t)arow1 * DIM + k0 + koffA);
    fa1.u[1] = *(const uint4*)(pw + (size_t)arow1 * DIM + k0 + 16 + koffA);
#pragma unroll
    for (int t = 0; t < 4; ++t) {
      int j = j0 + t * 16 + lrow;
      int n = j >> 8, p = j & 255;
      const __bf16* bp = outp + (((size_t)b * NN + n) * PIX + p) * DIM + k0 + kbB;
      FragB16 fb;
      fb.u[0] = *(const uint4*)(bp);
      fb.u[1] = *(const uint4*)(bp + 16);
      acc[0][t] = __builtin_amdgcn_wmma_f32_16x16x32_bf16(
          false, fa0.v, false, fb.v, (short)0, acc[0][t], false, false);
      acc[1][t] = __builtin_amdgcn_wmma_f32_16x16x32_bf16(
          false, fa1.v, false, fb.v, (short)0, acc[1][t], false, false);
    }
  }
#pragma unroll
  for (int mi = 0; mi < 2; ++mi) {
#pragma unroll
    for (int t = 0; t < 4; ++t) {
      int j = j0 + t * 16 + lrow;
      int n = j >> 8, p = j & 255;
      int hp = n / HP, wp = n % HP;
      int y  = hp * PP + (p >> 4);
      int x  = wp * PP + (p & 15);
#pragma unroll
      for (int v = 0; v < 8; ++v) {
        int oc = mgrp * 32 + mi * 16 + v + lhalf * 8;
        out[(((size_t)b * DIM + oc) * HH + y) * WW + x] = acc[mi][t][v] + pb[oc];
      }
    }
  }
}

// ---------------------------------------------------------------------------
// Launch
// ---------------------------------------------------------------------------
extern "C" void kernel_launch(void* const* d_in, const int* in_sizes, int n_in,
                              void* d_out, int out_size, void* d_ws, size_t ws_size,
                              hipStream_t stream) {
  const float* x       = (const float*)d_in[0];
  const float* patch_w = (const float*)d_in[1];
  const float* patch_b = (const float*)d_in[2];
  const float* qk_w    = (const float*)d_in[3];
  const float* v_w     = (const float*)d_in[4];
  const float* v_b     = (const float*)d_in[5];
  const float* proj_w  = (const float*)d_in[6];
  const float* proj_b  = (const float*)d_in[7];
  float* out = (float*)d_out;

  char* ws = (char*)d_ws;
  float*  xe   = (float*)(ws + OFF_XE);
  float*  q    = (float*)(ws + OFF_Q);
  float*  k    = (float*)(ws + OFF_K);
  __bf16* attn = (__bf16*)(ws + OFF_ATTN);
  __bf16* vbt  = (__bf16*)(ws + OFF_VBT);
  __bf16* outp = (__bf16*)(ws + OFF_OUTP);
  __bf16* pwb  = (__bf16*)(ws + OFF_PW);

  // front-end (small, fp32)
  k_patch_embed<<<(BB * NN * DIM + 255) / 256, 256, 0, stream>>>(x, patch_w, patch_b, xe);
  k_qk<<<(BB * NN * 2 * DIM + 255) / 256, 256, 0, stream>>>(xe, qk_w, q, k);
  k_attn_softmax<<<BB * HEADS * NP, 256, 0, stream>>>(q, k, attn);

  // V conv packed for WMMA B-operand
  {
    size_t total = (size_t)BB * DIM * PIX * NP;
    k_conv_v<<<(unsigned)((total + 255) / 256), 256, 0, stream>>>(x, v_w, v_b, vbt);
  }
  k_pack_w<<<(DIM * DIM + 255) / 256, 256, 0, stream>>>(proj_w, pwb);

  // GEMM-1: attn @ unfolded-V   (per (b,h): 224x8192x224, bf16 WMMA, 32x64 tiles)
  k_gemm_attn_v<<<dim3(BB * HEADS, 112), 256, 0, stream>>>(attn, vbt, outp);

  // GEMM-2: proj_w @ out_pre + bias, fold to NCHW fp32 (32x64 tiles)
  k_gemm_proj<<<dim3(BB, 784), 256, 0, stream>>>(pwb, outp, proj_b, out);
}
